// RGCNLayer_4758823764015
// MI455X (gfx1250) — matrix operand order repaired and checked by reading, not compiled, not run on hardware
//
#include <hip/hip_runtime.h>
#include <hip/hip_bf16.h>

// ---------------------------------------------------------------------------
// RGCN layer for MI455X (gfx1250, wave32, WMMA).
// Pipeline:
//   1) zero deg + h accumulator
//   2) deg[row] += edge_weight            (global_atomic_add_f32)
//   3) W_t[t][out][in] = sum_b alpha[t,b]*weight[b][in][out]  (bf16, transposed
//      so WMMA B-operand loads are contiguous); slot t=16 holds w_self^T
//   4) x -> bf16
//   5) h_all[t] = x @ W[t]  via v_wmma_f32_16x16x32_bf16 (bf16 out, 205MB ~ L2)
//      xs       = x @ w_self (fp32 out), same kernel, weight slot 16
//   6) per-edge: h[row] += norm_w * h_all[type, col]   (fp32 atomics)
//   7) per-node wave: LayerNorm(h)*gamma+beta + bias + xs -> out
// ---------------------------------------------------------------------------

typedef __attribute__((ext_vector_type(16))) __bf16 v16bf;
typedef __attribute__((ext_vector_type(8)))  __bf16 v8bf;
typedef __attribute__((ext_vector_type(4)))  __bf16 v4bf;
typedef __attribute__((ext_vector_type(8)))  float  v8f;

constexpr int NN = 50000;    // nodes
constexpr int EE = 1600000;  // edges
constexpr int FF = 128;      // in = out features
constexpr int TT = 16;       // edge types
constexpr int BB = 8;        // bases

__global__ void zero_f32(float* __restrict__ p, long long n) {
  long long i = (long long)blockIdx.x * blockDim.x + threadIdx.x;
  if (i < n) p[i] = 0.0f;
}

__global__ void deg_kernel(const int* __restrict__ row, const float* __restrict__ ew,
                           float* __restrict__ deg, int E) {
  int e = blockIdx.x * blockDim.x + threadIdx.x;
  if (e < E) unsafeAtomicAdd(&deg[row[e]], ew[e]);
}

// W_t[t][o][i]: t<16 -> basis combination, t==16 -> w_self transposed.
__global__ void combine_w(const float* __restrict__ alpha, const float* __restrict__ weight,
                          const float* __restrict__ w_self, __bf16* __restrict__ Wt) {
  int idx = blockIdx.x * blockDim.x + threadIdx.x;
  if (idx >= 17 * FF * FF) return;
  int t = idx >> 14;          // / (128*128)
  int o = (idx >> 7) & 127;
  int i = idx & 127;
  float s;
  if (t < TT) {
    s = 0.0f;
#pragma unroll
    for (int b = 0; b < BB; ++b)
      s += alpha[t * BB + b] * weight[((size_t)b * FF + i) * FF + o];
  } else {
    s = w_self[(size_t)i * FF + o];
  }
  Wt[((size_t)t * FF + o) * FF + i] = (__bf16)s;
}

__global__ void cvt_bf16(const float* __restrict__ x, __bf16* __restrict__ xb, int n) {
  int i = blockIdx.x * blockDim.x + threadIdx.x;
  if (i < n) xb[i] = (__bf16)x[i];
}

// C[t][m][0..127] = A[m][:] @ Bt[t][:][:]^T ; one wave = one 16x16 tile,
// 8 waves/block cover N=128, grid.x covers M/16, grid.z = weight slot.
template <typename OutT>
__global__ void __launch_bounds__(256)
gemm_bf16(const __bf16* __restrict__ A,   // [M][128] row-major
          const __bf16* __restrict__ Bt,  // [slots][128(out)][128(in)]
          OutT* __restrict__ C,           // [slots][M][128]
          int M) {
  const int wave = threadIdx.x >> 5;
  const int lane = threadIdx.x & 31;
  const int half = lane >> 4;    // K sub-range selector (ISA A/B layouts)
  const int l16  = lane & 15;    // row (A) / column (B) within tile
  const int m0 = blockIdx.x * 16;
  const int t  = blockIdx.z;
  const int n0 = wave * 16;

  const __bf16* Arow = A + (size_t)(m0 + l16) * FF;
  const __bf16* Bcol = Bt + ((size_t)t * FF + (n0 + l16)) * FF;

  v8f c = {};
#pragma unroll
  for (int k0 = 0; k0 < FF; k0 += 32) {
    // A 16x32 bf16 layout: lane m=l16, K = k0+16h .. k0+16h+15 (contiguous)
    v16bf a = *(const v16bf*)(Arow + k0 + 16 * half);
    // B 32x16 bf16 layout: lane n=l16, K = k0+16h .. k0+16h+15 (contiguous in W^T)
    v16bf b = *(const v16bf*)(Bcol + k0 + 16 * half);
    c = __builtin_amdgcn_wmma_f32_16x16x32_bf16(false, a, false, b,
                                                (short)0, c, false, false);
  }
  // C/D layout: VGPR r, lane l -> (m0 + r + 8*half, n0 + l16)
  OutT* Cb = C + ((size_t)t * M + m0) * FF + n0 + l16;
#pragma unroll
  for (int r = 0; r < 8; ++r)
    Cb[(size_t)(r + 8 * half) * FF] = (OutT)c[r];
}

// One wave per edge; lane handles 4 features. Gather bf16, scatter fp32 atomics.
__global__ void __launch_bounds__(256)
edge_scatter(const int* __restrict__ ei,      // [2][E]
             const int* __restrict__ et,      // [E]
             const float* __restrict__ ew,    // [E]
             const float* __restrict__ deg,   // [N]
             const __bf16* __restrict__ hall, // [T][N][128]
             float* __restrict__ h) {         // [N][128]
  int gw = (blockIdx.x * blockDim.x + threadIdx.x) >> 5;
  if (gw >= EE) return;
  int lane = threadIdx.x & 31;
  int r  = ei[gw];
  int cI = ei[EE + gw];
  int t  = et[gw];
  float nw = ew[gw] / fmaxf(deg[r], 1e-12f);
  const __bf16* src = hall + ((size_t)t * NN + cI) * FF + lane * 4;
  float* dst = h + (size_t)r * FF + lane * 4;
  v4bf v = *(const v4bf*)src;
#pragma unroll
  for (int i = 0; i < 4; ++i)
    unsafeAtomicAdd(dst + i, nw * (float)v[i]);
}

// One wave per node: LayerNorm + bias + self-loop term.
__global__ void __launch_bounds__(256)
finalize(const float* __restrict__ h, const float* __restrict__ xs,
         const float* __restrict__ gamma, const float* __restrict__ beta,
         const float* __restrict__ bias, float* __restrict__ out) {
  int n = (blockIdx.x * blockDim.x + threadIdx.x) >> 5;
  if (n >= NN) return;
  int lane = threadIdx.x & 31;
  int f = lane * 4;
  float4 v = *(const float4*)(h + (size_t)n * FF + f);

  float s = v.x + v.y + v.z + v.w;
#pragma unroll
  for (int m = 16; m >= 1; m >>= 1) s += __shfl_xor(s, m, 32);
  float mu = s * (1.0f / FF);

  float d0 = v.x - mu, d1 = v.y - mu, d2 = v.z - mu, d3 = v.w - mu;
  float q = d0 * d0 + d1 * d1 + d2 * d2 + d3 * d3;
#pragma unroll
  for (int m = 16; m >= 1; m >>= 1) q += __shfl_xor(q, m, 32);
  float inv = rsqrtf(q * (1.0f / FF) + 1e-5f);

  float4 xv = *(const float4*)(xs + (size_t)n * FF + f);
  float4 o;
  o.x = d0 * inv * gamma[f + 0] + beta[f + 0] + bias[f + 0] + xv.x;
  o.y = d1 * inv * gamma[f + 1] + beta[f + 1] + bias[f + 1] + xv.y;
  o.z = d2 * inv * gamma[f + 2] + beta[f + 2] + bias[f + 2] + xv.z;
  o.w = d3 * inv * gamma[f + 3] + beta[f + 3] + bias[f + 3] + xv.w;
  *(float4*)(out + (size_t)n * FF + f) = o;
}

extern "C" void kernel_launch(void* const* d_in, const int* in_sizes, int n_in,
                              void* d_out, int out_size, void* d_ws, size_t ws_size,
                              hipStream_t stream) {
  (void)in_sizes; (void)n_in; (void)out_size; (void)ws_size;
  const float* x      = (const float*)d_in[0];
  const int*   eidx   = (const int*)d_in[1];   // [2][E]
  const int*   etype  = (const int*)d_in[2];   // [E]
  const float* ew     = (const float*)d_in[3];
  const float* weight = (const float*)d_in[4]; // [B][128][128]
  const float* alpha  = (const float*)d_in[5]; // [T][B]
  const float* bias   = (const float*)d_in[6];
  const float* wself  = (const float*)d_in[7]; // [128][128]
  const float* gamma  = (const float*)d_in[8];
  const float* beta   = (const float*)d_in[9];
  float* out = (float*)d_out;

  char* ws = (char*)d_ws;
  size_t off = 0;
  auto take = [&](size_t bytes) -> char* {
    off = (off + 255) & ~(size_t)255;
    char* p = ws + off;
    off += bytes;
    return p;
  };
  float*  deg  = (float*)take((size_t)NN * 4);
  float*  h    = (float*)take((size_t)NN * FF * 4);
  float*  xs   = (float*)take((size_t)NN * FF * 4);
  __bf16* xb   = (__bf16*)take((size_t)NN * FF * 2);
  __bf16* Wt   = (__bf16*)take((size_t)17 * FF * FF * 2);
  __bf16* hall = (__bf16*)take((size_t)TT * NN * FF * 2);

  zero_f32<<<(NN + 255) / 256, 256, 0, stream>>>(deg, NN);
  zero_f32<<<((long long)NN * FF + 255) / 256, 256, 0, stream>>>(h, (long long)NN * FF);

  deg_kernel<<<(EE + 255) / 256, 256, 0, stream>>>(eidx, ew, deg, EE);

  combine_w<<<(17 * FF * FF + 255) / 256, 256, 0, stream>>>(alpha, weight, wself, Wt);
  cvt_bf16<<<(NN * FF + 255) / 256, 256, 0, stream>>>(x, xb, NN * FF);

  dim3 gT(NN / 16, 1, TT);
  gemm_bf16<__bf16><<<gT, 256, 0, stream>>>(xb, Wt, hall, NN);
  dim3 gS(NN / 16, 1, 1);
  gemm_bf16<float><<<gS, 256, 0, stream>>>(xb, Wt + (size_t)16 * FF * FF, xs, NN);

  edge_scatter<<<(EE * 32) / 256, 256, 0, stream>>>(eidx, etype, ew, deg, hall, h);

  finalize<<<(NN * 32 + 255) / 256, 256, 0, stream>>>(h, xs, gamma, beta, bias, out);
}
